// _GptOssAttention_85787676770793
// MI455X (gfx1250) — compile-verified, matrix operand order checked by reading, not run on hardware
//
#include <hip/hip_runtime.h>
#include <hip/hip_bf16.h>
#include <math.h>

// ---- problem constants (match reference) ----
#define HID   2880
#define NH    64
#define NKV   8
#define DH    64
#define SEQ   2048
#define QDIM  (NH * DH)    // 4096
#define KVDIM (NKV * DH)   // 512
#define SCALE 0.125f       // 64^-0.5
#define NEGB  -1e9f

typedef _Float16 h8_t  __attribute__((ext_vector_type(8)));
typedef _Float16 v16h  __attribute__((ext_vector_type(16)));
typedef float    v8f   __attribute__((ext_vector_type(8)));
typedef float    f4_t  __attribute__((ext_vector_type(4)));
typedef _Float16 h4_t  __attribute__((ext_vector_type(4)));

// ---------------------------------------------------------------------------
// Fragment loader for 16-bit WMMA A/B operands (CDNA5 16x16x32 layouts).
// A (16x32): lane 0-15 holds row M=lane, K = 0..7 & 16..23; lane 16-31 the
// same row with K = 8..15 & 24..31.  B (32x16) is symmetric with N per lane.
// Source stored row-major (row = M for A, = N for B^T) with contiguous K, so
// each half-fragment is one aligned 16-byte load.
// ---------------------------------------------------------------------------
static __device__ __forceinline__ v16h load_frag16(const _Float16* __restrict__ base,
                                                   int ld, int row, int kofs, int half) {
  const _Float16* p = base + (size_t)row * ld + kofs + half * 8;
  h8_t lo = *(const h8_t*)p;
  h8_t hi = *(const h8_t*)(p + 16);
  v16h r;
#pragma unroll
  for (int i = 0; i < 8; ++i) { r[i] = lo[i]; r[8 + i] = hi[i]; }
  return r;
}

static __device__ __forceinline__ v8f wmma_f16(v16h a, v16h b, v8f c) {
  return __builtin_amdgcn_wmma_f32_16x16x32_f16(false, a, false, b, (short)0, c, false, false);
}

// ---------------------------------------------------------------------------
// Vectorized f32 -> f16 convert (4 elements/thread)
// ---------------------------------------------------------------------------
__global__ void cvt_f16_kernel(const float* __restrict__ x, _Float16* __restrict__ y, size_t n4) {
  size_t i = (size_t)blockIdx.x * blockDim.x + threadIdx.x;
  if (i >= n4) return;
  f4_t v = *(const f4_t*)(x + i * 4);
  h4_t h;
#pragma unroll
  for (int j = 0; j < 4; ++j) h[j] = (_Float16)v[j];
  *(h4_t*)(y + i * 4) = h;
}

// ---------------------------------------------------------------------------
// LDS-tiled transpose-convert: W[I][O] f32 -> Wt[O][I] f16.
// 32x32 tile, coalesced global read and write; padded LDS row vs bank conflicts.
// Grid: (O/32, I/32), block (32, 8).
// ---------------------------------------------------------------------------
__global__ void transpose_cvt_kernel(const float* __restrict__ W, _Float16* __restrict__ Wt,
                                     int I, int O) {
  __shared__ float tile[32][33];
  int o0 = blockIdx.x * 32, i0 = blockIdx.y * 32;
  int tx = threadIdx.x, ty = threadIdx.y;
#pragma unroll
  for (int r = 0; r < 4; ++r)
    tile[ty + r * 8][tx] = W[(size_t)(i0 + ty + r * 8) * O + o0 + tx];
  __syncthreads();
#pragma unroll
  for (int r = 0; r < 4; ++r)
    Wt[(size_t)(o0 + ty + r * 8) * I + i0 + tx] = (_Float16)tile[tx][ty + r * 8];
}

// ---------------------------------------------------------------------------
// GEMM: C[M][N] = A[M][K] (f16) x Bt[N][K]^T (f16) + bias[N]
// Per-wave (BM*16) x (BN*16) tile, K stepped by 32 -> BM*BN WMMAs per step.
//   F32OUT : store f32 (final projection)  else f16
//   STORE_T: store C transposed f16 [N][M] (builds V^T)
// ---------------------------------------------------------------------------
template <int BM, int BN, bool F32OUT, bool STORE_T>
__global__ void gemm_kernel(const _Float16* __restrict__ A, const _Float16* __restrict__ Bt,
                            const float* __restrict__ bias, void* __restrict__ Cout,
                            int M, int N, int K) {
  int wid  = (int)((blockIdx.x * blockDim.x + threadIdx.x) >> 5);
  int lane = threadIdx.x & 31;
  int tn   = N / (BN * 16);
  int wm   = wid / tn, wn = wid % tn;
  if (wm >= M / (BM * 16)) return;
  int l15 = lane & 15, hl = (lane >> 4) & 1;

  v8f c[BM][BN] = {};
  int rA = wm * BM * 16 + l15;
  int rB = wn * BN * 16 + l15;
  for (int k = 0; k < K; k += 32) {
    v16h a[BM], b[BN];
#pragma unroll
    for (int mi = 0; mi < BM; ++mi) a[mi] = load_frag16(A, K, rA + mi * 16, k, hl);
#pragma unroll
    for (int ni = 0; ni < BN; ++ni) b[ni] = load_frag16(Bt, K, rB + ni * 16, k, hl);
#pragma unroll
    for (int mi = 0; mi < BM; ++mi)
#pragma unroll
      for (int ni = 0; ni < BN; ++ni)
        c[mi][ni] = wmma_f16(a[mi], b[ni], c[mi][ni]);
  }

#pragma unroll
  for (int mi = 0; mi < BM; ++mi)
#pragma unroll
    for (int ni = 0; ni < BN; ++ni) {
      int n = wn * BN * 16 + ni * 16 + l15;
      float bv = bias[n];
      int mbase = wm * BM * 16 + mi * 16 + 8 * hl;
      if (STORE_T) {                       // f16 [N][M], 8 contiguous m per lane
        h8_t v;
#pragma unroll
        for (int g = 0; g < 8; ++g) v[g] = (_Float16)(c[mi][ni][g] + bv);
        *(h8_t*)((_Float16*)Cout + (size_t)n * M + mbase) = v;
      } else if (F32OUT) {
#pragma unroll
        for (int g = 0; g < 8; ++g)
          ((float*)Cout)[(size_t)(mbase + g) * N + n] = c[mi][ni][g] + bv;
      } else {
#pragma unroll
        for (int g = 0; g < 8; ++g)
          ((_Float16*)Cout)[(size_t)(mbase + g) * N + n] = (_Float16)(c[mi][ni][g] + bv);
      }
    }
}

// ---------------------------------------------------------------------------
// Flash attention with sinks. One wave per (head, 16-query tile).
// Computes S^T = K x Q^T so softmax reduces along C-tile VGPRs (per-lane) and
// P^T re-packs into the B-fragment of O^T = V^T x P^T with no LDS/shuffles.
// Main loop: 64-kv blocks, NO causal compares (provably unmasked).
// Tail: 1-2 masked 32-kv blocks around the diagonal.
//   Q  : f16 [SEQ][QDIM]   Kf : f16 [SEQ][KVDIM]   Vt : f16 [KVDIM][SEQ]
//   O  : f16 [SEQ][QDIM]
// ---------------------------------------------------------------------------
__global__ void attn_kernel(const _Float16* __restrict__ Q, const _Float16* __restrict__ Kf,
                            const _Float16* __restrict__ Vt, const float* __restrict__ sinks,
                            _Float16* __restrict__ O) {
  int wid  = (int)((blockIdx.x * blockDim.x + threadIdx.x) >> 5);
  if (wid >= NH * (SEQ / 16)) return;
  int lane = threadIdx.x & 31, l15 = lane & 15, hl = (lane >> 4) & 1;
  int h  = wid >> 7;          // SEQ/16 = 128 q-tiles per head
  int qt = wid & 127;
  int kh = h >> 3;            // GQA: q head h -> kv head h/8
  int q0 = qt * 16;
  int q_idx = q0 + l15;

  const _Float16* Qh = Q  + h  * DH;
  const _Float16* Kh = Kf + kh * DH;
  const _Float16* Vh = Vt + (size_t)(kh * DH) * SEQ;

  v16h bq0 = load_frag16(Qh, QDIM, q0 + l15, 0,  hl);   // Q^T B-frags (d 0..63)
  v16h bq1 = load_frag16(Qh, QDIM, q0 + l15, 32, hl);

  v8f  o[4] = {};                                       // O^T: four 16x16 tiles
  float m_run = -__builtin_inff(), l_run = 0.f;

  // ---- main loop: fully-unmasked 64-kv blocks (kv <= 64*(q0/64)-1 < q0) ----
  int nfull = q0 >> 6;
  for (int j = 0; j < nfull; ++j) {
    int kb = j * 64;
    float p[4][8];
    float mblk = -__builtin_inff();
#pragma unroll
    for (int t = 0; t < 4; ++t) {
      int kv16 = kb + 16 * t;
      v16h ak0 = load_frag16(Kh, KVDIM, kv16 + l15, 0,  hl);
      v16h ak1 = load_frag16(Kh, KVDIM, kv16 + l15, 32, hl);
      v8f st = {};
      st = wmma_f16(ak0, bq0, st);
      st = wmma_f16(ak1, bq1, st);
#pragma unroll
      for (int g = 0; g < 8; ++g) {
        float s = st[g] * SCALE;
        p[t][g] = s;
        mblk = fmaxf(mblk, s);
      }
    }
    mblk = fmaxf(mblk, __shfl_xor(mblk, 16, 32));
    float m_new = fmaxf(m_run, mblk);
    float alpha = __expf(m_run - m_new);
    float lsum  = 0.f;
    v16h bp0, bp1;                                      // P^T B-frags (2x32 kv)
#pragma unroll
    for (int g = 0; g < 8; ++g) {
      float e0 = __expf(p[0][g] - m_new);
      float e1 = __expf(p[1][g] - m_new);
      float e2 = __expf(p[2][g] - m_new);
      float e3 = __expf(p[3][g] - m_new);
      lsum += (e0 + e1) + (e2 + e3);
      bp0[g] = (_Float16)e0; bp0[8 + g] = (_Float16)e1;
      bp1[g] = (_Float16)e2; bp1[8 + g] = (_Float16)e3;
    }
    lsum += __shfl_xor(lsum, 16, 32);
    l_run = l_run * alpha + lsum;
    m_run = m_new;
#pragma unroll
    for (int dt = 0; dt < 4; ++dt) {
#pragma unroll
      for (int g = 0; g < 8; ++g) o[dt][g] *= alpha;
      v16h av0 = load_frag16(Vh, SEQ, 16 * dt + l15, kb,      hl);
      v16h av1 = load_frag16(Vh, SEQ, 16 * dt + l15, kb + 32, hl);
      o[dt] = wmma_f16(av0, bp0, o[dt]);
      o[dt] = wmma_f16(av1, bp1, o[dt]);
    }
  }

  // ---- tail: masked 32-kv blocks covering kv in [64*nfull, q0+15] ----
  int kb0   = nfull * 64;
  int ntail = ((q0 & 63) + 47) >> 5;                    // 1 or 2
  for (int j = 0; j < ntail; ++j) {
    int kb = kb0 + j * 32;
    float p0[8], p1[8];
    float mblk = -__builtin_inff();
#pragma unroll
    for (int t = 0; t < 2; ++t) {
      int kv16 = kb + 16 * t;
      v16h ak0 = load_frag16(Kh, KVDIM, kv16 + l15, 0,  hl);
      v16h ak1 = load_frag16(Kh, KVDIM, kv16 + l15, 32, hl);
      v8f st = {};
      st = wmma_f16(ak0, bq0, st);
      st = wmma_f16(ak1, bq1, st);
      float* pp = t ? p1 : p0;
#pragma unroll
      for (int g = 0; g < 8; ++g) {
        int kv = kv16 + g + 8 * hl;                     // C-tile row = kv index
        float s = st[g] * SCALE + (kv <= q_idx ? 0.f : NEGB);
        pp[g] = s;
        mblk  = fmaxf(mblk, s);
      }
    }
    mblk = fmaxf(mblk, __shfl_xor(mblk, 16, 32));
    float m_new = fmaxf(m_run, mblk);
    float alpha = __expf(m_run - m_new);
    float lsum  = 0.f;
    v16h bp;
#pragma unroll
    for (int g = 0; g < 8; ++g) {
      float e0 = __expf(p0[g] - m_new);
      float e1 = __expf(p1[g] - m_new);
      lsum += e0 + e1;
      bp[g]     = (_Float16)e0;
      bp[8 + g] = (_Float16)e1;
    }
    lsum += __shfl_xor(lsum, 16, 32);
    l_run = l_run * alpha + lsum;
    m_run = m_new;
#pragma unroll
    for (int dt = 0; dt < 4; ++dt) {
#pragma unroll
      for (int g = 0; g < 8; ++g) o[dt][g] *= alpha;
      v16h av = load_frag16(Vh, SEQ, 16 * dt + l15, kb, hl);
      o[dt] = wmma_f16(av, bp, o[dt]);
    }
  }

  // learned sink: extra per-head logit in the softmax denominator
  float sk    = sinks[h];
  float mf    = fmaxf(m_run, sk);
  float fac   = __expf(m_run - mf);
  float denom = l_run * fac + __expf(sk - mf);
  float sc    = fac / denom;

#pragma unroll
  for (int dt = 0; dt < 4; ++dt) {
    h8_t v;
#pragma unroll
    for (int g = 0; g < 8; ++g) v[g] = (_Float16)(o[dt][g] * sc);
    // O^T element (d = 16*dt + g + 8*hl, q = l15): d contiguous over g
    *(h8_t*)(O + (size_t)(q0 + l15) * QDIM + h * DH + 16 * dt + 8 * hl) = v;
  }
}

// ---------------------------------------------------------------------------
extern "C" void kernel_launch(void* const* d_in, const int* in_sizes, int n_in,
                              void* d_out, int out_size, void* d_ws, size_t ws_size,
                              hipStream_t stream) {
  const float* hidden = (const float*)d_in[0];
  // d_in[1] = attention_bias (causal -1e9 mask) -> computed analytically
  const float* wq = (const float*)d_in[2];
  const float* bq = (const float*)d_in[3];
  const float* wk = (const float*)d_in[4];
  const float* bk = (const float*)d_in[5];
  const float* wv = (const float*)d_in[6];
  const float* bv = (const float*)d_in[7];
  const float* wo = (const float*)d_in[8];
  const float* bo = (const float*)d_in[9];
  const float* sinks = (const float*)d_in[10];
  float* out = (float*)d_out;

  // workspace carve-up (f16 buffers), 256B aligned
  size_t off = 0;
  auto carve = [&](size_t elems) { size_t r = off; off = (off + elems * 2 + 255) & ~(size_t)255; return r; };
  char* ws = (char*)d_ws;
  _Float16* Xh  = (_Float16*)(ws + carve((size_t)SEQ * HID));
  _Float16* Wqt = (_Float16*)(ws + carve((size_t)QDIM * HID));
  _Float16* Wkt = (_Float16*)(ws + carve((size_t)KVDIM * HID));
  _Float16* Wvt = (_Float16*)(ws + carve((size_t)KVDIM * HID));
  _Float16* Wot = (_Float16*)(ws + carve((size_t)HID * QDIM));
  _Float16* Qf  = (_Float16*)(ws + carve((size_t)SEQ * QDIM));
  _Float16* Kf  = (_Float16*)(ws + carve((size_t)SEQ * KVDIM));
  _Float16* Vtb = (_Float16*)(ws + carve((size_t)KVDIM * SEQ));
  _Float16* At  = (_Float16*)(ws + carve((size_t)SEQ * QDIM));
  (void)ws_size; (void)in_sizes; (void)n_in; (void)out_size;

  const int TPB = 256;

  // 1. convert hidden (vectorized) + LDS-tiled weight transposes to f16
  cvt_f16_kernel<<<(unsigned)(((size_t)SEQ * HID / 4 + TPB - 1) / TPB), TPB, 0, stream>>>(
      hidden, Xh, (size_t)SEQ * HID / 4);
  dim3 tb(32, 8);
  transpose_cvt_kernel<<<dim3(QDIM / 32, HID / 32),  tb, 0, stream>>>(wq, Wqt, HID, QDIM);
  transpose_cvt_kernel<<<dim3(KVDIM / 32, HID / 32), tb, 0, stream>>>(wk, Wkt, HID, KVDIM);
  transpose_cvt_kernel<<<dim3(KVDIM / 32, HID / 32), tb, 0, stream>>>(wv, Wvt, HID, KVDIM);
  transpose_cvt_kernel<<<dim3(HID / 32, QDIM / 32),  tb, 0, stream>>>(wo, Wot, QDIM, HID);

  // 2. QKV projections (V stored transposed -> V^T)
  auto grid_waves = [&](size_t waves) { return (unsigned)((waves * 32 + TPB - 1) / TPB); };
  gemm_kernel<4, 4, false, false><<<grid_waves((SEQ / 64) * (QDIM / 64)), TPB, 0, stream>>>(
      Xh, Wqt, bq, Qf, SEQ, QDIM, HID);
  gemm_kernel<2, 2, false, false><<<grid_waves((SEQ / 32) * (KVDIM / 32)), TPB, 0, stream>>>(
      Xh, Wkt, bk, Kf, SEQ, KVDIM, HID);
  gemm_kernel<2, 2, false, true><<<grid_waves((SEQ / 32) * (KVDIM / 32)), TPB, 0, stream>>>(
      Xh, Wvt, bv, Vtb, SEQ, KVDIM, HID);

  // 3. flash attention with sinks: 64 heads x 128 q-tiles = 8192 waves
  attn_kernel<<<grid_waves(NH * (SEQ / 16)), TPB, 0, stream>>>(Qf, Kf, Vtb, sinks, At);

  // 4. output projection -> f32 d_out
  gemm_kernel<4, 4, true, false><<<grid_waves((SEQ / 64) * (HID / 64)), TPB, 0, stream>>>(
      At, Wot, bo, out, SEQ, HID, QDIM);
}
